// AttentionWithRoPE_28166395528055
// MI455X (gfx1250) — compile-verified, matrix operand order checked by reading, not run on hardware
//
#include <hip/hip_runtime.h>

// ---------------------------------------------------------------------------
// AttentionWithRoPE on MI455X (gfx1250): bf16 WMMA pipeline
//   1) cvt hidden -> bf16 ; cvt+transpose Wq/Wk/Wv/Wo -> bf16 [n][k]
//   2) gemm128 (mode 0/1): H x W -> Q/K with fused RoPE, write (B,H,S,D) bf16
//      gemm128 (mode 2):   H x Wv -> V,              write (B,H,S,D) bf16
//   3) flash_attn: causal online-softmax attention, write O (B,S,H*D) bf16
//   4) gemm128 (mode 3): O x Wo -> f32 d_out
// Staging uses CDNA5 async global->LDS (ASYNCcnt) with LDS double buffering.
// ---------------------------------------------------------------------------

typedef __attribute__((ext_vector_type(16))) __bf16 v16bf;
typedef __attribute__((ext_vector_type(8)))  float  v8f;

#define EMBED 2048
#define NH    16
#define HD    128
#define NBATCH 2
#define SEQ   2048
#define MROWS (NBATCH * SEQ)   // 4096

union Frag {
    v16bf v;
    uint4 q[2];
};

__device__ __forceinline__ v8f wmma_bf16(const Frag& a, const Frag& b, v8f c) {
    return __builtin_amdgcn_wmma_f32_16x16x32_bf16(
        /*neg_a=*/false, a.v, /*neg_b=*/false, b.v,
        /*c_mod=*/(short)0, c, /*reuse_a=*/false, /*reuse_b=*/false);
}

// flat address of a __shared__ object keeps the LDS byte offset in its low 32b
__device__ __forceinline__ unsigned lds_addr32(const void* p) {
    return (unsigned)(unsigned long long)p;
}

__device__ __forceinline__ void s_wait_async0() {
    asm volatile("s_wait_asynccnt 0x0" ::: "memory");
}

// async copy of 64 contiguous bytes (per lane) global -> LDS
__device__ __forceinline__ void async_copy64(void* lds_dst, const void* gsrc) {
    unsigned l = lds_addr32(lds_dst);
    asm volatile(
        "global_load_async_to_lds_b128 %0, %1, off\n\t"
        "global_load_async_to_lds_b128 %0, %1, off offset:16\n\t"
        "global_load_async_to_lds_b128 %0, %1, off offset:32\n\t"
        "global_load_async_to_lds_b128 %0, %1, off offset:48"
        :: "v"(l), "v"(gsrc) : "memory");
}

// async copy of 128 contiguous bytes (per lane) global -> LDS
__device__ __forceinline__ void async_copy128(void* lds_dst, const void* gsrc) {
    unsigned l = lds_addr32(lds_dst);
    asm volatile(
        "global_load_async_to_lds_b128 %0, %1, off\n\t"
        "global_load_async_to_lds_b128 %0, %1, off offset:16\n\t"
        "global_load_async_to_lds_b128 %0, %1, off offset:32\n\t"
        "global_load_async_to_lds_b128 %0, %1, off offset:48\n\t"
        "global_load_async_to_lds_b128 %0, %1, off offset:64\n\t"
        "global_load_async_to_lds_b128 %0, %1, off offset:80\n\t"
        "global_load_async_to_lds_b128 %0, %1, off offset:96\n\t"
        "global_load_async_to_lds_b128 %0, %1, off offset:112"
        :: "v"(l), "v"(gsrc) : "memory");
}

// ---------------------------------------------------------------------------
// elementwise f32 -> bf16
// ---------------------------------------------------------------------------
__global__ void cvt_bf16_kernel(const float* __restrict__ in,
                                __bf16* __restrict__ out, long n) {
    long i = (long)blockIdx.x * blockDim.x + threadIdx.x;
    long stride = (long)gridDim.x * blockDim.x;
    for (; i < n; i += stride) out[i] = (__bf16)in[i];
}

// ---------------------------------------------------------------------------
// W (2048x2048 f32, [k][n]) -> Wt (bf16, [n][k])
// ---------------------------------------------------------------------------
__global__ __launch_bounds__(256)
void transpose_cvt_kernel(const float* __restrict__ W, __bf16* __restrict__ Wt) {
    __shared__ float tile[32][33];
    int bx = blockIdx.x * 32;   // n range
    int by = blockIdx.y * 32;   // k range
    int tx = threadIdx.x & 31;
    int ty = threadIdx.x >> 5;  // 0..7
    #pragma unroll
    for (int j = 0; j < 32; j += 8)
        tile[ty + j][tx] = W[(size_t)(by + ty + j) * EMBED + bx + tx];
    __syncthreads();
    #pragma unroll
    for (int j = 0; j < 32; j += 8)
        Wt[(size_t)(bx + ty + j) * EMBED + by + tx] = (__bf16)tile[tx][ty + j];
}

// ---------------------------------------------------------------------------
// GEMM: C(128x128 tile) = A(M x 2048) * Bt^T, Bt stored [n][k] row-major bf16.
// mode 0: Q proj + RoPE -> out_bf (B,H,S,D)
// mode 1: K proj + RoPE -> out_bf (B,H,S,D)
// mode 2: V proj        -> out_bf (B,H,S,D)
// mode 3: O proj        -> out_f32 (B*S, 2048)
// grid: (16 n-tiles/heads, 32 m-tiles), block 256 (8 waves)
// Double-buffered async global->LDS staging.
// ---------------------------------------------------------------------------
__global__ __launch_bounds__(256)
void gemm128_kernel(const __bf16* __restrict__ A, const __bf16* __restrict__ Bt,
                    const float* __restrict__ cosp, const float* __restrict__ sinp,
                    __bf16* __restrict__ out_bf, float* __restrict__ out_f32,
                    int mode) {
    __shared__ __align__(16) float smem[128 * 128];          // 64KB epilogue buf
    __shared__ __align__(16) __bf16 stage[2][2][128 * 32];   // [buf][a/b] 32KB

    const int tid   = threadIdx.x;
    const int wave  = tid >> 5;
    const int lane  = tid & 31;
    const int lhalf = lane & 15;
    const int hi    = lane >> 4;
    const int k0    = hi ? 8 : 0;          // A/B 16-bit fragment k-origin

    const int m0 = blockIdx.y * 128;
    const int n0 = blockIdx.x * 128;
    const int nlocal = wave * 16;

    v8f acc[8];
    #pragma unroll
    for (int m = 0; m < 8; ++m) acc[m] = (v8f)0.0f;

    // per-thread staging source/dest (threads 0..127 -> A, 128..255 -> Bt)
    const int r = tid & 127;
    const __bf16* srow = (tid < 128) ? (A  + (size_t)(m0 + r) * EMBED)
                                     : (Bt + (size_t)(n0 + r) * EMBED);
    const int part = (tid < 128) ? 0 : 1;

    // prologue: stage k=0 into buffer 0
    async_copy64(&stage[0][part][r * 32], srow);
    s_wait_async0();
    __syncthreads();

    int cur = 0;
    for (int kk = 0; kk < EMBED; kk += 32) {
        // kick off next tile into the other buffer (no wait yet)
        if (kk + 32 < EMBED) {
            async_copy64(&stage[cur ^ 1][part][r * 32], srow + kk + 32);
            __builtin_prefetch(srow + kk + 64, 0, 3);
        }

        // compute from current buffer
        const __bf16* sa = stage[cur][0];
        const __bf16* sb = stage[cur][1];
        Frag bfrag, afrag[8];
        bfrag.q[0] = *(const uint4*)(sb + (nlocal + lhalf) * 32 + k0);
        bfrag.q[1] = *(const uint4*)(sb + (nlocal + lhalf) * 32 + k0 + 16);
        #pragma unroll
        for (int m = 0; m < 8; ++m) {
            afrag[m].q[0] = *(const uint4*)(sa + (m * 16 + lhalf) * 32 + k0);
            afrag[m].q[1] = *(const uint4*)(sa + (m * 16 + lhalf) * 32 + k0 + 16);
        }
        #pragma unroll
        for (int m = 0; m < 8; ++m)
            acc[m] = wmma_bf16(afrag[m], bfrag, acc[m]);

        s_wait_async0();
        __syncthreads();
        cur ^= 1;
    }

    // spill C tile (f32) to LDS for cross-column epilogue (RoPE needs d +/- 64)
    #pragma unroll
    for (int m = 0; m < 8; ++m)
        #pragma unroll
        for (int rr = 0; rr < 8; ++rr)
            smem[(m * 16 + rr + hi * 8) * 128 + (nlocal + lhalf)] = acc[m][rr];
    __syncthreads();

    const int h = blockIdx.x;   // head index for modes 0..2
    #pragma unroll 4
    for (int i = 0; i < 64; ++i) {
        int idx = tid + 256 * i;            // 0..16383
        int row = idx >> 7;
        int d   = idx & 127;
        int gr  = m0 + row;                 // global row = b*SEQ + s
        int b   = gr >> 11;
        int s   = gr & (SEQ - 1);
        float x = smem[row * 128 + d];
        if (mode <= 1) {   // RoPE
            float xr = (d < 64) ? -smem[row * 128 + d + 64]
                                :  smem[row * 128 + d - 64];
            float c  = cosp[((size_t)b * SEQ + s) * HD + d];
            float sn = sinp[((size_t)b * SEQ + s) * HD + d];
            x = x * c + xr * sn;
        }
        if (mode <= 2) {   // (B,H,S,D) bf16
            out_bf[(((size_t)(b * NH + h)) * SEQ + s) * HD + d] = (__bf16)x;
        } else {           // f32 final output
            out_f32[(size_t)gr * EMBED + n0 + d] = x;
        }
    }
}

// ---------------------------------------------------------------------------
// Flash attention, causal. grid: (B*H = 32, q-tiles = 16), block 256 (8 waves)
// q,k,v: (B,H,S,D) bf16.  o: (B,S,H*D) bf16.
// ---------------------------------------------------------------------------
__global__ __launch_bounds__(256)
void flash_attn_kernel(const __bf16* __restrict__ q, const __bf16* __restrict__ k,
                       const __bf16* __restrict__ v, __bf16* __restrict__ o) {
    __shared__ __align__(16) __bf16 kt[128 * 128];        // [key][d]   32KB
    __shared__ __align__(16) __bf16 vt[128 * 128];        // [d][key]   32KB
    __shared__ __align__(16) __bf16 pls[8 * 16 * 128];    // per-wave P 32KB

    const int tid   = threadIdx.x;
    const int wave  = tid >> 5;
    const int lane  = tid & 31;
    const int lhalf = lane & 15;
    const int hi    = lane >> 4;
    const int k0    = hi ? 8 : 0;

    const int bh = blockIdx.x;
    const int q0 = blockIdx.y * 128;
    const __bf16* qbase = q + (size_t)bh * SEQ * HD;
    const __bf16* kbase = k + (size_t)bh * SEQ * HD;
    const __bf16* vbase = v + (size_t)bh * SEQ * HD;

    // Q fragments (A layout), 4 chunks over D=128
    Frag qf[4];
    {
        const __bf16* qp = qbase + (size_t)(q0 + wave * 16 + lhalf) * HD;
        #pragma unroll
        for (int c = 0; c < 4; ++c) {
            qf[c].q[0] = *(const uint4*)(qp + c * 32 + k0);
            qf[c].q[1] = *(const uint4*)(qp + c * 32 + k0 + 16);
        }
    }

    v8f oacc[8];
    float rmax[8], rsum[8];
    #pragma unroll
    for (int t = 0; t < 8; ++t) oacc[t] = (v8f)0.0f;
    #pragma unroll
    for (int rr = 0; rr < 8; ++rr) { rmax[rr] = -3.0e38f; rsum[rr] = 0.0f; }

    __bf16* pw = pls + wave * 16 * 128;
    const float scale = 0.088388347648318447f;   // 1/sqrt(128)
    const int kbmax = q0 >> 7;

    for (int kb = 0; kb <= kbmax; ++kb) {
        __syncthreads();
        {   // stage K tile [key][d] via async global->LDS (128B per thread)
            int rr = tid >> 1;
            int hh = tid & 1;
            async_copy128(kt + rr * 128 + hh * 64,
                          kbase + (size_t)(kb * 128 + rr) * HD + hh * 64);
        }
        // stage V tile transposed -> [d][key] (register path: needs transpose)
        #pragma unroll
        for (int i = 0; i < 8; ++i) {
            int idx = tid + 256 * i;          // 0..2047
            int key = idx >> 4;
            int d0  = (idx & 15) * 8;
            union { uint4 q4; __bf16 e[8]; } u;
            u.q4 = *(const uint4*)(vbase + (size_t)(kb * 128 + key) * HD + d0);
            #pragma unroll
            for (int j = 0; j < 8; ++j) vt[(d0 + j) * 128 + key] = u.e[j];
        }
        s_wait_async0();
        __syncthreads();

        // scores: 8 tiles of 16 keys, each = 4 WMMAs over D
        v8f sc[8];
        #pragma unroll
        for (int t8 = 0; t8 < 8; ++t8) {
            v8f s_ = (v8f)0.0f;
            const int krow = t8 * 16 + lhalf;
            #pragma unroll
            for (int c = 0; c < 4; ++c) {
                Frag kf;
                kf.q[0] = *(const uint4*)(kt + krow * 128 + c * 32 + k0);
                kf.q[1] = *(const uint4*)(kt + krow * 128 + c * 32 + k0 + 16);
                s_ = wmma_bf16(qf[c], kf, s_);
            }
            sc[t8] = s_;
        }

        // scale + causal mask
        #pragma unroll
        for (int t8 = 0; t8 < 8; ++t8) {
            int keyg = kb * 128 + t8 * 16 + lhalf;
            #pragma unroll
            for (int rr = 0; rr < 8; ++rr) {
                float sv = sc[t8][rr] * scale;
                int qg = q0 + wave * 16 + rr + hi * 8;
                if (keyg > qg) sv = -3.0e38f;
                sc[t8][rr] = sv;
            }
        }

        // row max over tiles, then across the 16 lanes of this half-wave
        float lmax[8];
        #pragma unroll
        for (int rr = 0; rr < 8; ++rr) {
            float m = sc[0][rr];
            #pragma unroll
            for (int t8 = 1; t8 < 8; ++t8) m = fmaxf(m, sc[t8][rr]);
            lmax[rr] = m;
        }
        #pragma unroll
        for (int off = 1; off < 16; off <<= 1)
            #pragma unroll
            for (int rr = 0; rr < 8; ++rr)
                lmax[rr] = fmaxf(lmax[rr], __shfl_xor(lmax[rr], off, 32));

        float fac[8];
        #pragma unroll
        for (int rr = 0; rr < 8; ++rr) {
            float nm = fmaxf(rmax[rr], lmax[rr]);
            fac[rr] = __expf(rmax[rr] - nm);
            rmax[rr] = nm;
        }
        #pragma unroll
        for (int t = 0; t < 8; ++t)
            #pragma unroll
            for (int rr = 0; rr < 8; ++rr) oacc[t][rr] *= fac[rr];

        // P = exp(s - m), partial sums
        float ls[8];
        #pragma unroll
        for (int rr = 0; rr < 8; ++rr) ls[rr] = 0.0f;
        #pragma unroll
        for (int t8 = 0; t8 < 8; ++t8)
            #pragma unroll
            for (int rr = 0; rr < 8; ++rr) {
                float p = __expf(sc[t8][rr] - rmax[rr]);
                sc[t8][rr] = p;
                ls[rr] += p;
            }
        #pragma unroll
        for (int off = 1; off < 16; off <<= 1)
            #pragma unroll
            for (int rr = 0; rr < 8; ++rr) ls[rr] += __shfl_xor(ls[rr], off, 32);
        #pragma unroll
        for (int rr = 0; rr < 8; ++rr) rsum[rr] = rsum[rr] * fac[rr] + ls[rr];

        // C-layout -> A-layout via per-wave LDS scratch (wave LDS ops in order)
        #pragma unroll
        for (int t8 = 0; t8 < 8; ++t8)
            #pragma unroll
            for (int rr = 0; rr < 8; ++rr)
                pw[(rr + hi * 8) * 128 + t8 * 16 + lhalf] = (__bf16)sc[t8][rr];

        // O += P * V : 4 key-chunks x 8 d-tiles
        #pragma unroll
        for (int c = 0; c < 4; ++c) {
            Frag pf;
            pf.q[0] = *(const uint4*)(pw + lhalf * 128 + c * 32 + k0);
            pf.q[1] = *(const uint4*)(pw + lhalf * 128 + c * 32 + k0 + 16);
            #pragma unroll
            for (int t = 0; t < 8; ++t) {
                Frag vf;
                vf.q[0] = *(const uint4*)(vt + (t * 16 + lhalf) * 128 + c * 32 + k0);
                vf.q[1] = *(const uint4*)(vt + (t * 16 + lhalf) * 128 + c * 32 + k0 + 16);
                oacc[t] = wmma_bf16(pf, vf, oacc[t]);
            }
        }
    }

    // finalize: o[(b*SEQ+s)*2048 + h*128 + d]
    const int b = bh >> 4, h = bh & 15;
    #pragma unroll
    for (int t = 0; t < 8; ++t)
        #pragma unroll
        for (int rr = 0; rr < 8; ++rr) {
            int s = q0 + wave * 16 + rr + hi * 8;
            int d = t * 16 + lhalf;
            float val = oacc[t][rr] / rsum[rr];
            o[((size_t)(b * SEQ + s)) * EMBED + h * HD + d] = (__bf16)val;
        }
}

// ---------------------------------------------------------------------------
extern "C" void kernel_launch(void* const* d_in, const int* in_sizes, int n_in,
                              void* d_out, int out_size, void* d_ws, size_t ws_size,
                              hipStream_t stream) {
    const float* hs   = (const float*)d_in[0];
    const float* cosp = (const float*)d_in[1];
    const float* sinp = (const float*)d_in[2];
    const float* Wq   = (const float*)d_in[3];
    const float* Wk   = (const float*)d_in[4];
    const float* Wv   = (const float*)d_in[5];
    const float* Wo   = (const float*)d_in[6];
    float* out = (float*)d_out;

    size_t off = 0;
    char* base = (char*)d_ws;
    auto alloc = [&](size_t bytes) -> char* {
        char* p = base + off;
        off += (bytes + 255) & ~(size_t)255;
        return p;
    };
    const size_t ACT = (size_t)MROWS * EMBED * sizeof(__bf16);   // 16 MB
    const size_t WB  = (size_t)EMBED * EMBED * sizeof(__bf16);   //  8 MB
    __bf16* h_bf = (__bf16*)alloc(ACT);
    __bf16* wq_t = (__bf16*)alloc(WB);
    __bf16* wk_t = (__bf16*)alloc(WB);
    __bf16* wv_t = (__bf16*)alloc(WB);
    __bf16* wo_t = (__bf16*)alloc(WB);
    __bf16* q_ws = (__bf16*)alloc(ACT);
    __bf16* k_ws = (__bf16*)alloc(ACT);
    __bf16* v_ws = (__bf16*)alloc(ACT);
    __bf16* o_ws = (__bf16*)alloc(ACT);

    // 1) conversions
    {
        long n = (long)MROWS * EMBED;
        cvt_bf16_kernel<<<4096, 256, 0, stream>>>(hs, h_bf, n);
        dim3 tg(EMBED / 32, EMBED / 32);
        transpose_cvt_kernel<<<tg, 256, 0, stream>>>(Wq, wq_t);
        transpose_cvt_kernel<<<tg, 256, 0, stream>>>(Wk, wk_t);
        transpose_cvt_kernel<<<tg, 256, 0, stream>>>(Wv, wv_t);
        transpose_cvt_kernel<<<tg, 256, 0, stream>>>(Wo, wo_t);
    }

    // 2) Q/K/V projections (N-tile == one head), RoPE fused for Q,K
    dim3 gg(EMBED / 128, MROWS / 128);   // (16, 32)
    gemm128_kernel<<<gg, 256, 0, stream>>>(h_bf, wq_t, cosp, sinp, q_ws, nullptr, 0);
    gemm128_kernel<<<gg, 256, 0, stream>>>(h_bf, wk_t, cosp, sinp, k_ws, nullptr, 1);
    gemm128_kernel<<<gg, 256, 0, stream>>>(h_bf, wv_t, cosp, sinp, v_ws, nullptr, 2);

    // 3) causal flash attention
    dim3 ga(NBATCH * NH, SEQ / 128);     // (32, 16)
    flash_attn_kernel<<<ga, 256, 0, stream>>>(q_ws, k_ws, v_ws, o_ws);

    // 4) output projection -> f32 d_out
    gemm128_kernel<<<gg, 256, 0, stream>>>(o_ws, wo_t, nullptr, nullptr, nullptr, out, 3);
}
